// QuantizedMultiHeadCrossAttention_12171937317142
// MI455X (gfx1250) — 8-bit matrix operand order repaired and checked by reading, compiled, not run on hardware
//
#include <hip/hip_runtime.h>
#include <stdint.h>

typedef __attribute__((ext_vector_type(2))) float v2f;
typedef __attribute__((ext_vector_type(8))) float v8f;
typedef __attribute__((ext_vector_type(8))) int   v8i;

#define D_MODEL   2048
#define NUM_HEADS 16
#define HEAD_DIM  128
#define NQ        4096
#define NK        1024
#define ATTN_SCALE 0.08838834764831845f

__global__ __launch_bounds__(256) void gemm_bias_f32_wmma(
    const float* __restrict__ A, const float* __restrict__ W,
    const float* __restrict__ bias, float* __restrict__ C,
    int M, int N, int K)
{
  const int lane = threadIdx.x & 31;
  const int wid  = blockIdx.x * (blockDim.x >> 5) + (threadIdx.x >> 5);
  const int tilesN = N >> 6;
  const int tm = wid / tilesN;
  const int tn = wid - tm * tilesN;
  if (tm >= (M >> 4)) return;
  const int row0 = tm << 4;
  const int col0 = tn << 6;
  const int m    = lane & 15;
  const int half = lane >> 4;

  v8f acc[4];
#pragma unroll
  for (int j = 0; j < 4; ++j)
#pragma unroll
    for (int r = 0; r < 8; ++r) acc[j][r] = 0.0f;

  const float* arow = A + (size_t)(row0 + m) * K + (half << 1);
  for (int k0 = 0; k0 < K; k0 += 4) {
    v2f a = *(const v2f*)(arow + k0);
    const float* w0 = W + (size_t)(k0 + (half << 1)) * N + col0 + m;
#pragma unroll
    for (int j = 0; j < 4; ++j) {
      v2f b;
      b[0] = w0[(j << 4)];
      b[1] = w0[(j << 4) + N];
      acc[j] = __builtin_amdgcn_wmma_f32_16x16x4_f32(
          false, a, false, b, (short)0, acc[j], false, false);
    }
  }

#pragma unroll
  for (int j = 0; j < 4; ++j) {
    const int col = col0 + (j << 4) + m;
    const float bv = bias[col];
#pragma unroll
    for (int r = 0; r < 8; ++r) {
      const int row = row0 + (half << 3) + r;
      C[(size_t)row * N + col] = acc[j][r] + bv;
    }
  }
}

__device__ __forceinline__ int q_sym127(float x, float inv) {
  float r = rintf(x * inv);
  r = fminf(fmaxf(r, -127.0f), 127.0f);
  return (int)r;
}

__global__ __launch_bounds__(256) void quant_rows_i8(
    const float* __restrict__ src, int rowStride, int R, float premul,
    int8_t* __restrict__ out8, float* __restrict__ outScale)
{
  const int lane = threadIdx.x & 31;
  const int wid  = blockIdx.x * (blockDim.x >> 5) + (threadIdx.x >> 5);
  const int h    = wid & (NUM_HEADS - 1);
  const int row  = wid >> 4;
  if (row >= R) return;

  const float* p = src + (size_t)row * rowStride + h * HEAD_DIM + (lane << 2);
  float x0 = p[0] * premul, x1 = p[1] * premul, x2 = p[2] * premul, x3 = p[3] * premul;
  float amax = fmaxf(fmaxf(fabsf(x0), fabsf(x1)), fmaxf(fabsf(x2), fabsf(x3)));
#pragma unroll
  for (int off = 16; off > 0; off >>= 1)
    amax = fmaxf(amax, __shfl_xor(amax, off, 32));
  const float s   = fmaxf(amax * (1.0f / 127.0f), 1e-8f);
  const float inv = 1.0f / s;

  uint32_t packed = ((uint32_t)(q_sym127(x0, inv) & 255))        |
                    ((uint32_t)(q_sym127(x1, inv) & 255) << 8)   |
                    ((uint32_t)(q_sym127(x2, inv) & 255) << 16)  |
                    ((uint32_t)(q_sym127(x3, inv) & 255) << 24);
  *(uint32_t*)(out8 + (size_t)(h * R + row) * HEAD_DIM + (lane << 2)) = packed;
  if (lane == 0) outScale[h * R + row] = s;
}

__global__ __launch_bounds__(256) void quant_v_cols_i8(
    const float* __restrict__ kv, int8_t* __restrict__ v8out,
    float* __restrict__ vScale)
{
  const int lane = threadIdx.x & 31;
  const int c    = blockIdx.x * (blockDim.x >> 5) + (threadIdx.x >> 5);
  if (c >= D_MODEL) return;

  const float* base = kv + D_MODEL + c;
  const int m0 = lane << 5;
  float vals[32];
  float amax = 0.0f;
#pragma unroll
  for (int i = 0; i < 32; ++i) {
    float v = base[(size_t)(m0 + i) * (2 * D_MODEL)];
    vals[i] = v;
    amax = fmaxf(amax, fabsf(v));
  }
#pragma unroll
  for (int off = 16; off > 0; off >>= 1)
    amax = fmaxf(amax, __shfl_xor(amax, off, 32));
  const float s   = fmaxf(amax * (1.0f / 127.0f), 1e-8f);
  const float inv = 1.0f / s;

  int8_t* outc = v8out + (size_t)c * NK + m0;
#pragma unroll
  for (int i = 0; i < 32; i += 4) {
    uint32_t packed = ((uint32_t)(q_sym127(vals[i + 0], inv) & 255))       |
                      ((uint32_t)(q_sym127(vals[i + 1], inv) & 255) << 8)  |
                      ((uint32_t)(q_sym127(vals[i + 2], inv) & 255) << 16) |
                      ((uint32_t)(q_sym127(vals[i + 3], inv) & 255) << 24);
    *(uint32_t*)(outc + i) = packed;
  }
  if (lane == 0) vScale[c] = s;
}

#define ATTN_SMEM (16 * NK * 4 + 16 * NK + 64 * 4)

__global__ __launch_bounds__(128) void attn_i8_wmma(
    const int8_t* __restrict__ q8, const float* __restrict__ qs,
    const int8_t* __restrict__ k8, const float* __restrict__ ks,
    const int8_t* __restrict__ v8q, const float* __restrict__ vs,
    float* __restrict__ O)
{
  extern __shared__ char smem[];
  float*   S   = (float*)smem;
  uint8_t* A8  = (uint8_t*)(smem + 16 * NK * 4);
  float*   red = (float*)(smem + 16 * NK * 4 + 16 * NK);

  const int h    = blockIdx.x >> 8;
  const int qt   = blockIdx.x & 255;
  const int lane = threadIdx.x & 31;
  const int wave = threadIdx.x >> 5;
  const int m    = lane & 15;
  const int half = lane >> 4;

  if (threadIdx.x < 16)
    red[48 + threadIdx.x] = qs[h * NQ + qt * 16 + threadIdx.x];

  const int8_t* qrow = q8 + (size_t)(h * NQ + qt * 16 + m) * HEAD_DIM;
  v8i aq0, aq1;
#pragma unroll
  for (int v = 0; v < 8; ++v) {
    const int off = ((v & 3) << 2) + (half << 4) + ((v >> 2) << 5);
    aq0[v] = *(const int*)(qrow + off);
    aq1[v] = *(const int*)(qrow + 64 + off);
  }
  __syncthreads();

  for (int kt = wave; kt < NK / 16; kt += 4) {
    const int key = kt * 16 + m;
    const int8_t* krow = k8 + (size_t)(h * NK + key) * HEAD_DIM;
    v8i b0, b1;
#pragma unroll
    for (int v = 0; v < 8; ++v) {
      const int off = ((v & 3) << 2) + (half << 4) + ((v >> 2) << 5);
      b0[v] = *(const int*)(krow + off);
      b1[v] = *(const int*)(krow + 64 + off);
    }
    v8i acc = {0, 0, 0, 0, 0, 0, 0, 0};
    acc = __builtin_amdgcn_wmma_i32_16x16x64_iu8(true, aq0, true, b0, acc, false, false);
    acc = __builtin_amdgcn_wmma_i32_16x16x64_iu8(true, aq1, true, b1, acc, false, false);
    const float kscale = ks[h * NK + key];
#pragma unroll
    for (int r = 0; r < 8; ++r) {
      const int mr = (half << 3) + r;
      S[mr * NK + key] = (float)acc[r] * red[48 + mr] * kscale;
    }
  }
  __syncthreads();

  {
    const int row = threadIdx.x >> 3;
    const int sub = threadIdx.x & 7;
    const float* srow = S + row * NK;
    float mx = -3.0e38f;
    for (int j = sub; j < NK; j += 8) mx = fmaxf(mx, srow[j]);
#pragma unroll
    for (int off = 1; off < 8; off <<= 1) mx = fmaxf(mx, __shfl_xor(mx, off, 32));
    float sum = 0.0f;
    for (int j = sub; j < NK; j += 8) sum += __expf(srow[j] - mx);
#pragma unroll
    for (int off = 1; off < 8; off <<= 1) sum += __shfl_xor(sum, off, 32);
    if (sub == 0) {
      red[row]      = mx;
      red[32 + row] = 1.0f / (255.0f * sum);
    }
  }
  __syncthreads();

  for (int idx = threadIdx.x; idx < 16 * NK; idx += 128) {
    const int row = idx >> 10;
    float r = rintf(255.0f * __expf(S[idx] - red[row]));
    r = fminf(r, 255.0f);
    A8[idx] = (uint8_t)(int)r;
  }
  __syncthreads();

  for (int dt = wave; dt < 8; dt += 4) {
    const int d = dt * 16 + m;
    const int8_t* vcol = v8q + (size_t)(h * HEAD_DIM + d) * NK;
    const float vscale = vs[h * HEAD_DIM + d];
    v8i acc = {0, 0, 0, 0, 0, 0, 0, 0};
    for (int kc = 0; kc < NK / 64; ++kc) {
      const uint8_t* arow = A8 + m * NK + kc * 64;
      const int8_t*  bcol = vcol + kc * 64;
      v8i a, b;
#pragma unroll
      for (int v = 0; v < 8; ++v) {
        const int aoff = ((v & 3) << 2) + (half << 4) + ((v >> 2) << 5);
        const int boff = ((v & 3) << 2) + (half << 4) + ((v >> 2) << 5);
        a[v] = *(const int*)(arow + aoff);
        b[v] = *(const int*)(bcol + boff);
      }
      acc = __builtin_amdgcn_wmma_i32_16x16x64_iu8(false, a, true, b, acc, false, false);
    }
#pragma unroll
    for (int r = 0; r < 8; ++r) {
      const int mr  = (half << 3) + r;
      const int row = qt * 16 + mr;
      O[(size_t)row * D_MODEL + h * HEAD_DIM + d] = (float)acc[r] * red[32 + mr] * vscale;
    }
  }
}

extern "C" void kernel_launch(void* const* d_in, const int* in_sizes, int n_in,
                              void* d_out, int out_size, void* d_ws, size_t ws_size,
                              hipStream_t stream) {
  const float* x    = (const float*)d_in[0];
  const float* cond = (const float*)d_in[1];
  const float* Wq   = (const float*)d_in[2];
  const float* bq   = (const float*)d_in[3];
  const float* Wkv  = (const float*)d_in[4];
  const float* bkv  = (const float*)d_in[5];
  const float* Wp   = (const float*)d_in[6];
  const float* bp   = (const float*)d_in[7];
  float* out = (float*)d_out;

  size_t off = 0;
  auto take = [&](size_t bytes) -> void* {
    void* p = (char*)d_ws + off;
    off += (bytes + 255) & ~(size_t)255;
    return p;
  };
  float*  qbuf  = (float*)take((size_t)NQ * D_MODEL * 4);
  float*  kvbuf = (float*)take((size_t)NK * 2 * D_MODEL * 4);
  int8_t* q8    = (int8_t*)take((size_t)NUM_HEADS * NQ * HEAD_DIM);
  int8_t* k8    = (int8_t*)take((size_t)NUM_HEADS * NK * HEAD_DIM);
  int8_t* v8q   = (int8_t*)take((size_t)D_MODEL * NK);
  float*  qsb   = (float*)take((size_t)NUM_HEADS * NQ * 4);
  float*  ksb   = (float*)take((size_t)NUM_HEADS * NK * 4);
  float*  vsb   = (float*)take((size_t)D_MODEL * 4);
  float*  Obuf  = qbuf;

  gemm_bias_f32_wmma<<<(NQ / 16) * (D_MODEL / 64) / 8, 256, 0, stream>>>(
      x, Wq, bq, qbuf, NQ, D_MODEL, D_MODEL);
  gemm_bias_f32_wmma<<<(NK / 16) * (2 * D_MODEL / 64) / 8, 256, 0, stream>>>(
      cond, Wkv, bkv, kvbuf, NK, 2 * D_MODEL, D_MODEL);

  quant_rows_i8<<<NQ * NUM_HEADS / 8, 256, 0, stream>>>(
      qbuf, D_MODEL, NQ, ATTN_SCALE, q8, qsb);
  quant_rows_i8<<<NK * NUM_HEADS / 8, 256, 0, stream>>>(
      kvbuf, 2 * D_MODEL, NK, 1.0f, k8, ksb);
  quant_v_cols_i8<<<D_MODEL / 8, 256, 0, stream>>>(kvbuf, v8q, vsb);

  attn_i8_wmma<<<NUM_HEADS * (NQ / 16), 128, ATTN_SMEM, stream>>>(
      q8, qsb, k8, ksb, v8q, vsb, Obuf);

  gemm_bias_f32_wmma<<<(NQ / 16) * (D_MODEL / 64) / 8, 256, 0, stream>>>(
      Obuf, Wp, bp, out, NQ, D_MODEL, D_MODEL);
}
